// Attention3D_1503238553896
// MI455X (gfx1250) — compile-verified, matrix operand order checked by reading, not run on hardware
//
#include <hip/hip_runtime.h>

#define B_      2
#define C_      256
#define N_      32768
#define H_      4
#define HD_     64
#define OC3_    768
#define GROUPS_ 8
#define CPG_    32
#define EPS_    1e-5f
#define GN_SPLIT 64
#define GN_CHUNK ((CPG_ * N_) / GN_SPLIT)   // 16384
#define SK_CHUNK 1024
#define SK_SPLIT (N_ / SK_CHUNK)            // 32

typedef __attribute__((ext_vector_type(16))) __bf16 v16bf;
typedef __attribute__((ext_vector_type(8)))  __bf16 v8bf;
typedef __attribute__((ext_vector_type(8)))  float  v8f;

// float -> bf16 bits, round-to-nearest-even
static __device__ __forceinline__ unsigned short f2bf(float f) {
    union { float f; unsigned u; } x; x.f = f;
    unsigned r = (x.u + 0x7fffu + ((x.u >> 16) & 1u)) >> 16;
    return (unsigned short)r;
}

// Load one WMMA 16x16x32 bf16 A/B fragment for this lane.
// p points at (row/col base) + kb where kb = (lane>=16)*8 elements.
// Fragment = 8 bf16 at p (K kb..kb+7) and 8 bf16 at p+16 (K kb+16..kb+23).
static __device__ __forceinline__ v16bf frag_ld(const unsigned short* p) {
    union { v16bf v; v8bf h[2]; } u;
    u.h[0] = *(const v8bf*)(p);
    u.h[1] = *(const v8bf*)(p + 16);
    return u.v;
}

static __device__ __forceinline__ v8f wmma_bf16(v16bf a, v16bf b, v8f c) {
    return __builtin_amdgcn_wmma_f32_16x16x32_bf16(false, a, false, b, (short)0, c,
                                                   false, false);
}

// ---------------- GroupNorm: split partial reduction ----------------
__global__ void k_gn_partial(const float* __restrict__ x, float* __restrict__ part) {
    int s = blockIdx.x % GN_SPLIT;
    int g = (blockIdx.x / GN_SPLIT) % GROUPS_;
    int b = blockIdx.x / (GN_SPLIT * GROUPS_);
    size_t base = ((size_t)b * C_ + (size_t)g * CPG_) * N_ + (size_t)s * GN_CHUNK;
    float s1 = 0.f, s2 = 0.f;
    for (int i = threadIdx.x; i < GN_CHUNK; i += 256) {
        float v = x[base + i];
        s1 += v; s2 += v * v;
    }
    __shared__ float r1[256], r2[256];
    r1[threadIdx.x] = s1; r2[threadIdx.x] = s2;
    __syncthreads();
    for (int off = 128; off > 0; off >>= 1) {
        if ((int)threadIdx.x < off) {
            r1[threadIdx.x] += r1[threadIdx.x + off];
            r2[threadIdx.x] += r2[threadIdx.x + off];
        }
        __syncthreads();
    }
    if (threadIdx.x == 0) {
        part[blockIdx.x * 2 + 0] = r1[0];
        part[blockIdx.x * 2 + 1] = r2[0];
    }
}

__global__ void k_gn_finalize(const float* __restrict__ part, float* __restrict__ stats) {
    int t = threadIdx.x;                       // 32 threads = 1 wave
    int base = blockIdx.x * GN_SPLIT;
    float s1 = part[(base + t) * 2]     + part[(base + t + 32) * 2];
    float s2 = part[(base + t) * 2 + 1] + part[(base + t + 32) * 2 + 1];
    for (int off = 16; off > 0; off >>= 1) {
        s1 += __shfl_xor(s1, off);
        s2 += __shfl_xor(s2, off);
    }
    if (t == 0) {
        const float cnt = (float)CPG_ * (float)N_;
        float mean = s1 / cnt;
        float var  = s2 / cnt - mean * mean;
        stats[blockIdx.x * 2 + 0] = mean;
        stats[blockIdx.x * 2 + 1] = rsqrtf(var + EPS_);
    }
}

// ---------------- weight f32 -> bf16 ----------------
__global__ void k_convw(const float* __restrict__ wq, const float* __restrict__ wp,
                        unsigned short* __restrict__ wqh, unsigned short* __restrict__ wph) {
    int i = blockIdx.x * 256 + threadIdx.x;
    if (i < OC3_ * C_) wqh[i] = f2bf(wq[i]);
    if (i < C_ * C_)   wph[i] = f2bf(wp[i]);
}

// ---------------- normalize + transpose to xn_t (b, n, c) bf16 ----------------
__global__ void k_gnorm_t(const float* __restrict__ x, const float* __restrict__ gw,
                          const float* __restrict__ gb, const float* __restrict__ stats,
                          unsigned short* __restrict__ xnt) {
    int b  = blockIdx.z;
    int c0 = blockIdx.y * 32;                  // one group per block (CPG_==32)
    int n0 = blockIdx.x * 64;
    int g  = blockIdx.y;
    float mean = stats[(b * GROUPS_ + g) * 2 + 0];
    float rstd = stats[(b * GROUPS_ + g) * 2 + 1];
    __shared__ unsigned short tile[32][72];    // padded vs bank conflicts
    int t = threadIdx.x;
    int nl = t & 63, cl0 = t >> 6;
    #pragma unroll
    for (int r = 0; r < 8; ++r) {
        int cl = cl0 + 4 * r;
        float v = x[((size_t)b * C_ + c0 + cl) * N_ + n0 + nl];
        float y = (v - mean) * rstd * gw[c0 + cl] + gb[c0 + cl];
        tile[cl][nl] = f2bf(y);
    }
    __syncthreads();
    int cl = t & 31, nl0 = t >> 5;
    #pragma unroll
    for (int r = 0; r < 8; ++r) {
        int nn = nl0 * 8 + r;
        xnt[((size_t)b * N_ + n0 + nn) * C_ + c0 + cl] = tile[cl][nn];
    }
}

// ---------------- qkv GEMM: (768x256) x (256xN) -> scatter q/k/v ----------------
// 8 waves: 2(M) x 4(N), each wave owns a 32x32 quadrant (2x2 wmma tiles).
// q,k: (b,h,hd,n) bf16.  v: transposed (b,h,n,hd) bf16.
__global__ void k_qkv(const unsigned short* __restrict__ wq,
                      const float* __restrict__ bias,
                      const unsigned short* __restrict__ xnt,
                      unsigned short* __restrict__ q,
                      unsigned short* __restrict__ k,
                      unsigned short* __restrict__ vt) {
    int nblk = blockIdx.x, mblk = blockIdx.y, b = blockIdx.z;
    int lane = threadIdx.x & 31, wid = threadIdx.x >> 5;
    int wm = wid & 1, wn = wid >> 1;           // 2 M-waves x 4 N-waves
    int m0 = mblk * 64 + wm * 32;
    int n0 = nblk * 128 + wn * 32;
    int l15 = lane & 15, kb = (lane >> 4) * 8;
    const unsigned short* ap0 = wq  + (size_t)(m0 + l15) * C_ + kb;
    const unsigned short* ap1 = ap0 + (size_t)16 * C_;
    const unsigned short* bp0 = xnt + ((size_t)b * N_ + n0 + l15) * C_ + kb;
    const unsigned short* bp1 = bp0 + (size_t)16 * C_;
    v8f acc00 = {}, acc01 = {}, acc10 = {}, acc11 = {};
    #pragma unroll
    for (int kk = 0; kk < C_; kk += 32) {
        v16bf a0 = frag_ld(ap0 + kk);
        v16bf a1 = frag_ld(ap1 + kk);
        v16bf b0 = frag_ld(bp0 + kk);
        v16bf b1 = frag_ld(bp1 + kk);
        acc00 = wmma_bf16(a0, b0, acc00);
        acc01 = wmma_bf16(a0, b1, acc01);
        acc10 = wmma_bf16(a1, b0, acc10);
        acc11 = wmma_bf16(a1, b1, acc11);
    }
    // mblk*64 rows lie entirely within one of q/k/v and one head
    int part = mblk >> 2;                      // 0:q 1:k 2:v
    int head = mblk & 3;
    int half = lane >> 4;
    v8f accs[2][2] = {{acc00, acc01}, {acc10, acc11}};
    if (part < 2) {
        unsigned short* dst = (part ? k : q) + (size_t)(b * H_ + head) * HD_ * N_;
        #pragma unroll
        for (int ti = 0; ti < 2; ++ti) {
            #pragma unroll
            for (int tj = 0; tj < 2; ++tj) {
                int n = n0 + tj * 16 + l15;
                #pragma unroll
                for (int j = 0; j < 8; ++j) {
                    int r = wm * 32 + ti * 16 + half * 8 + j;
                    dst[(size_t)r * N_ + n] =
                        f2bf(accs[ti][tj][j] + bias[part * 256 + head * 64 + r]);
                }
            }
        }
    } else {
        #pragma unroll
        for (int ti = 0; ti < 2; ++ti) {
            int rbase = wm * 32 + ti * 16 + half * 8;
            #pragma unroll
            for (int tj = 0; tj < 2; ++tj) {
                int n = n0 + tj * 16 + l15;
                unsigned short pk[8];
                #pragma unroll
                for (int j = 0; j < 8; ++j)
                    pk[j] = f2bf(accs[ti][tj][j] + bias[512 + head * 64 + rbase + j]);
                uint4 val;
                val.x = pk[0] | ((unsigned)pk[1] << 16);
                val.y = pk[2] | ((unsigned)pk[3] << 16);
                val.z = pk[4] | ((unsigned)pk[5] << 16);
                val.w = pk[6] | ((unsigned)pk[7] << 16);
                *(uint4*)(vt + ((size_t)(b * H_ + head) * N_ + n) * HD_ + rbase) = val;
            }
        }
    }
}

// ---------------- S = q . k^T, split-K over n (deterministic partials) ----------------
// 4 waves (128 threads): 2(M) x 2(N), each wave a 32x32 quadrant of the 64x64 output.
__global__ void k_qk(const unsigned short* __restrict__ q,
                     const unsigned short* __restrict__ kk,
                     float* __restrict__ Sp) {
    int bh = blockIdx.y;
    int kc = blockIdx.x * SK_CHUNK;
    int lane = threadIdx.x & 31, wid = threadIdx.x >> 5;
    int wm = wid & 1, wn = wid >> 1;
    int l15 = lane & 15, kb = (lane >> 4) * 8;
    const unsigned short* ap0 = q  + ((size_t)bh * HD_ + wm * 32 + l15) * N_ + kc + kb;
    const unsigned short* ap1 = ap0 + (size_t)16 * N_;
    const unsigned short* bp0 = kk + ((size_t)bh * HD_ + wn * 32 + l15) * N_ + kc + kb;
    const unsigned short* bp1 = bp0 + (size_t)16 * N_;
    v8f acc00 = {}, acc01 = {}, acc10 = {}, acc11 = {};
    for (int t = 0; t < SK_CHUNK; t += 32) {
        if ((t & 127) == 0) {                  // uniform branch; prefetch 512B ahead
            __builtin_prefetch(ap0 + t + 256, 0, 0);
            __builtin_prefetch(ap1 + t + 256, 0, 0);
            __builtin_prefetch(bp0 + t + 256, 0, 0);
            __builtin_prefetch(bp1 + t + 256, 0, 0);
        }
        v16bf a0 = frag_ld(ap0 + t);
        v16bf a1 = frag_ld(ap1 + t);
        v16bf b0 = frag_ld(bp0 + t);
        v16bf b1 = frag_ld(bp1 + t);
        acc00 = wmma_bf16(a0, b0, acc00);
        acc01 = wmma_bf16(a0, b1, acc01);
        acc10 = wmma_bf16(a1, b0, acc10);
        acc11 = wmma_bf16(a1, b1, acc11);
    }
    int half = lane >> 4;
    float* out = Sp + ((size_t)blockIdx.x * (B_ * H_) + bh) * 64 * 64;
    v8f accs[2][2] = {{acc00, acc01}, {acc10, acc11}};
    #pragma unroll
    for (int ti = 0; ti < 2; ++ti) {
        #pragma unroll
        for (int tj = 0; tj < 2; ++tj) {
            int ncol = wn * 32 + tj * 16 + l15;
            #pragma unroll
            for (int j = 0; j < 8; ++j) {
                int m = wm * 32 + ti * 16 + half * 8 + j;
                out[m * 64 + ncol] = accs[ti][tj][j];
            }
        }
    }
}

__global__ void k_sred(const float* __restrict__ Sp, float* __restrict__ S) {
    int i = blockIdx.x * 256 + threadIdx.x;    // < B*H*64*64 = 32768
    float s = 0.f;
    for (int sp = 0; sp < SK_SPLIT; ++sp)
        s += Sp[(size_t)sp * (B_ * H_ * 64 * 64) + i];
    S[i] = s;
}

// ---------------- softmax over 64 cols; one wave per row; out bf16 ----------------
__global__ void k_softmax(const float* __restrict__ S, unsigned short* __restrict__ P) {
    int row = blockIdx.x;
    int t = threadIdx.x;                       // 32 threads
    const float* s = S + (size_t)row * 64;
    float v0 = s[t] * 0.125f;                  // hd^-0.5 = 64^-0.5
    float v1 = s[t + 32] * 0.125f;
    float mx = fmaxf(v0, v1);
    for (int off = 16; off > 0; off >>= 1) mx = fmaxf(mx, __shfl_xor(mx, off));
    float e0 = __expf(v0 - mx), e1 = __expf(v1 - mx);
    float sm = e0 + e1;
    for (int off = 16; off > 0; off >>= 1) sm += __shfl_xor(sm, off);
    float inv = 1.f / sm;
    P[(size_t)row * 64 + t]      = f2bf(e0 * inv);
    P[(size_t)row * 64 + t + 32] = f2bf(e1 * inv);
}

// ---------------- O_t(b,n,c) = P(64x64) . v(64xN), written transposed ----------------
// 4 waves: 2(M) x 2(N) quadrants of the 64(M) x 64(N) block tile.
__global__ void k_av(const unsigned short* __restrict__ P,
                     const unsigned short* __restrict__ vt,
                     unsigned short* __restrict__ Ot) {
    int bh = blockIdx.y;
    int nblk = blockIdx.x;
    int b = bh >> 2, head = bh & 3;
    int lane = threadIdx.x & 31, wid = threadIdx.x >> 5;
    int wm = wid & 1, wn = wid >> 1;
    int l15 = lane & 15, kb = (lane >> 4) * 8;
    int n0 = nblk * 64;
    const unsigned short* ap0 = P  + ((size_t)bh * HD_ + wm * 32 + l15) * HD_ + kb;
    const unsigned short* ap1 = ap0 + (size_t)16 * HD_;
    const unsigned short* bp0 = vt + ((size_t)bh * N_ + n0 + wn * 32 + l15) * HD_ + kb;
    const unsigned short* bp1 = bp0 + (size_t)16 * HD_;
    v8f acc00 = {}, acc01 = {}, acc10 = {}, acc11 = {};
    #pragma unroll
    for (int t = 0; t < HD_; t += 32) {
        v16bf a0 = frag_ld(ap0 + t);
        v16bf a1 = frag_ld(ap1 + t);
        v16bf b0 = frag_ld(bp0 + t);
        v16bf b1 = frag_ld(bp1 + t);
        acc00 = wmma_bf16(a0, b0, acc00);
        acc01 = wmma_bf16(a0, b1, acc01);
        acc10 = wmma_bf16(a1, b0, acc10);
        acc11 = wmma_bf16(a1, b1, acc11);
    }
    int half = lane >> 4;
    v8f accs[2][2] = {{acc00, acc01}, {acc10, acc11}};
    #pragma unroll
    for (int ti = 0; ti < 2; ++ti) {
        int cbase = head * 64 + wm * 32 + ti * 16 + half * 8;
        #pragma unroll
        for (int tj = 0; tj < 2; ++tj) {
            int n = n0 + wn * 32 + tj * 16 + l15;
            unsigned short pk[8];
            #pragma unroll
            for (int j = 0; j < 8; ++j) pk[j] = f2bf(accs[ti][tj][j]);
            uint4 val;
            val.x = pk[0] | ((unsigned)pk[1] << 16);
            val.y = pk[2] | ((unsigned)pk[3] << 16);
            val.z = pk[4] | ((unsigned)pk[5] << 16);
            val.w = pk[6] | ((unsigned)pk[7] << 16);
            *(uint4*)(Ot + ((size_t)b * N_ + n) * C_ + cbase) = val;
        }
    }
}

// ---------------- proj GEMM + bias + residual, f32 out ----------------
// 8 waves: 2(M) x 4(N), 2x2 tiles per wave, block tile 64 x 128.
__global__ void k_proj(const unsigned short* __restrict__ wp,
                       const float* __restrict__ bias,
                       const unsigned short* __restrict__ Ot,
                       const float* __restrict__ x,
                       float* __restrict__ out) {
    int nblk = blockIdx.x, mblk = blockIdx.y, b = blockIdx.z;
    int lane = threadIdx.x & 31, wid = threadIdx.x >> 5;
    int wm = wid & 1, wn = wid >> 1;
    int m0 = mblk * 64 + wm * 32;
    int n0 = nblk * 128 + wn * 32;
    int l15 = lane & 15, kb = (lane >> 4) * 8;
    const unsigned short* ap0 = wp + (size_t)(m0 + l15) * C_ + kb;
    const unsigned short* ap1 = ap0 + (size_t)16 * C_;
    const unsigned short* bp0 = Ot + ((size_t)b * N_ + n0 + l15) * C_ + kb;
    const unsigned short* bp1 = bp0 + (size_t)16 * C_;
    v8f acc00 = {}, acc01 = {}, acc10 = {}, acc11 = {};
    #pragma unroll
    for (int t = 0; t < C_; t += 32) {
        v16bf a0 = frag_ld(ap0 + t);
        v16bf a1 = frag_ld(ap1 + t);
        v16bf b0 = frag_ld(bp0 + t);
        v16bf b1 = frag_ld(bp1 + t);
        acc00 = wmma_bf16(a0, b0, acc00);
        acc01 = wmma_bf16(a0, b1, acc01);
        acc10 = wmma_bf16(a1, b0, acc10);
        acc11 = wmma_bf16(a1, b1, acc11);
    }
    int half = lane >> 4;
    v8f accs[2][2] = {{acc00, acc01}, {acc10, acc11}};
    #pragma unroll
    for (int ti = 0; ti < 2; ++ti) {
        #pragma unroll
        for (int tj = 0; tj < 2; ++tj) {
            int n = n0 + tj * 16 + l15;
            #pragma unroll
            for (int j = 0; j < 8; ++j) {
                int o = m0 + ti * 16 + half * 8 + j;
                size_t idx = ((size_t)b * C_ + o) * N_ + n;
                out[idx] = x[idx] + bias[o] + accs[ti][tj][j];
            }
        }
    }
}

extern "C" void kernel_launch(void* const* d_in, const int* in_sizes, int n_in,
                              void* d_out, int out_size, void* d_ws, size_t ws_size,
                              hipStream_t stream) {
    const float* x      = (const float*)d_in[0];
    const float* gn_w   = (const float*)d_in[1];
    const float* gn_b   = (const float*)d_in[2];
    const float* qkv_w  = (const float*)d_in[3];
    const float* qkv_b  = (const float*)d_in[4];
    const float* proj_w = (const float*)d_in[5];
    const float* proj_b = (const float*)d_in[6];
    float* out = (float*)d_out;

    char* w = (char*)d_ws;
    size_t off = 0;
    auto alloc = [&](size_t bytes) -> void* {
        void* p = w + off;
        off = (off + bytes + 255) & ~(size_t)255;
        return p;
    };
    float* part  = (float*)alloc((size_t)B_ * GROUPS_ * GN_SPLIT * 2 * sizeof(float));
    float* stats = (float*)alloc((size_t)B_ * GROUPS_ * 2 * sizeof(float));
    unsigned short* wqh = (unsigned short*)alloc((size_t)OC3_ * C_ * 2);
    unsigned short* wph = (unsigned short*)alloc((size_t)C_ * C_ * 2);
    unsigned short* xnt = (unsigned short*)alloc((size_t)B_ * N_ * C_ * 2);
    unsigned short* qb  = (unsigned short*)alloc((size_t)B_ * H_ * HD_ * N_ * 2);
    unsigned short* kbf = (unsigned short*)alloc((size_t)B_ * H_ * HD_ * N_ * 2);
    unsigned short* vtb = (unsigned short*)alloc((size_t)B_ * H_ * N_ * HD_ * 2);
    float* Sp = (float*)alloc((size_t)SK_SPLIT * B_ * H_ * 64 * 64 * sizeof(float));
    float* Sb = (float*)alloc((size_t)B_ * H_ * 64 * 64 * sizeof(float));
    unsigned short* Pb = (unsigned short*)alloc((size_t)B_ * H_ * 64 * 64 * 2);
    unsigned short* Ot = (unsigned short*)alloc((size_t)B_ * N_ * C_ * 2);

    k_gn_partial<<<B_ * GROUPS_ * GN_SPLIT, 256, 0, stream>>>(x, part);
    k_gn_finalize<<<B_ * GROUPS_, 32, 0, stream>>>(part, stats);
    k_convw<<<(OC3_ * C_) / 256, 256, 0, stream>>>(qkv_w, proj_w, wqh, wph);
    k_gnorm_t<<<dim3(N_ / 64, C_ / 32, B_), 256, 0, stream>>>(x, gn_w, gn_b, stats, xnt);
    k_qkv<<<dim3(N_ / 128, OC3_ / 64, B_), 256, 0, stream>>>(wqh, qkv_b, xnt, qb, kbf, vtb);
    k_qk<<<dim3(SK_SPLIT, B_ * H_), 128, 0, stream>>>(qb, kbf, Sp);
    k_sred<<<(B_ * H_ * 64 * 64) / 256, 256, 0, stream>>>(Sp, Sb);
    k_softmax<<<B_ * H_ * 64, 32, 0, stream>>>(Sb, Pb);
    k_av<<<dim3(N_ / 64, B_ * H_), 128, 0, stream>>>(Pb, vtb, Ot);
    k_proj<<<dim3(N_ / 128, C_ / 64, B_), 256, 0, stream>>>(wph, proj_b, Ot, x, out);
}
